// Y_decoder_12137577578918
// MI455X (gfx1250) — compile-verified, matrix operand order checked by reading, not run on hardware
//
#include <hip/hip_runtime.h>
#include <hip/hip_bf16.h>
#include <stdint.h>

#define N_NODES 100000
#define N_EDGES 1600000
#define NUM_FEATS 128
#define LATENT 64
#define IN1 (NUM_FEATS + LATENT)   /* 192 = 6 * 32  (exact K tiling) */
#define HIDDEN 128                 /* 8 * 16        (exact N tiling) */
#define BN_EPS 1e-5f

#define KSTEPS 6                   /* 192 / 32 */
#define NTILES 8                   /* 128 / 16 */
#define MT_PER_BLOCK 10            /* 6250 = 625 * 10 */
/* packed W1: [nTile][kstep][lane][16 bf16]  -> 8*6*32*16 bf16 = 49152 B */
#define WPACK_ELEMS (NTILES * KSTEPS * 32 * 16)

typedef __attribute__((ext_vector_type(16))) __bf16 v16bf;
typedef __attribute__((ext_vector_type(8)))  __bf16 v8bf;
typedef __attribute__((ext_vector_type(8)))  float  v8f;

__device__ __forceinline__ float atomAddF(float* p, float v) {
  return __hip_atomic_fetch_add(p, v, __ATOMIC_RELAXED, __HIP_MEMORY_SCOPE_AGENT);
}

__device__ __forceinline__ int kpair_base(int v, int g) {
  // ISA 7.12.2 16-bit A/B operand: VGPR v holds K pair (v<4 ? 2v : 16+2(v-4)) + 8g
  return ((v < 4) ? (2 * v) : (16 + 2 * (v - 4))) + 8 * g;
}

// ---------------------------------------------------------------- degrees ---
__global__ __launch_bounds__(256) void k_degree(const long long* __restrict__ dst,
                                                unsigned* __restrict__ deg) {
  int e = blockIdx.x * blockDim.x + threadIdx.x;
  if (e < N_EDGES) atomicAdd(&deg[(int)dst[e]], 1u);
}

__global__ __launch_bounds__(256) void k_dinv(const unsigned* __restrict__ deg,
                                              float* __restrict__ dinv) {
  int i = blockIdx.x * blockDim.x + threadIdx.x;
  if (i < N_NODES) dinv[i] = rsqrtf((float)(deg[i] + 1u));  // +1 self loop
}

// ------------------------- pack A = concat(u_Y, X) to row-major bf16 -------
// Fully coalesced streaming pass; removes all f32->bf16 cvt from the GEMM.
__global__ __launch_bounds__(256) void k_packA(const float* __restrict__ uY,
                                               const float* __restrict__ X,
                                               __bf16* __restrict__ Ap) {
  int idx = blockIdx.x * blockDim.x + threadIdx.x;
  const int total = N_NODES * IN1;
  if (idx >= total) return;
  int i = idx / IN1, k = idx % IN1;
  float v = (k < LATENT) ? uY[(size_t)i * LATENT + k]
                         : X[(size_t)i * NUM_FEATS + (k - LATENT)];
  Ap[idx] = (__bf16)v;
}

// ---------------------------------------- pack W1 into WMMA B-operand order -
// Wp[((nTile*6 + kstep)*32 + lane)*16 + 2v(+1)] = bf16(W1[kb(+1)][nTile*16+ml])
__global__ __launch_bounds__(256) void k_packW1(const float* __restrict__ W1,
                                                __bf16* __restrict__ Wp) {
  int t = blockIdx.x * blockDim.x + threadIdx.x;   // 0 .. 8*6*32-1
  if (t >= NTILES * KSTEPS * 32) return;
  int lane  = t & 31;
  int kstep = (t >> 5) % KSTEPS;
  int nTile = t / (KSTEPS * 32);
  int g = lane >> 4, ml = lane & 15;
  int col = nTile * 16 + ml;
  __bf16* dst = Wp + (size_t)t * 16;
#pragma unroll
  for (int v = 0; v < 8; ++v) {
    int kb = kpair_base(v, g) + kstep * 32;
    dst[2 * v]     = (__bf16)W1[(size_t)kb * HIDDEN + col];
    dst[2 * v + 1] = (__bf16)W1[(size_t)(kb + 1) * HIDDEN + col];
  }
}

// ------------------------------------------------------- GEMM1 via WMMA ----
// hs1[i,f] = dinv[i] * ( concat(u_Y, X)[i,:] @ W1[:,f] )
// Block = 8 waves; wave w owns nTile = w; block covers 10 consecutive mTiles.
// Packed W1 (49 KB) staged to LDS via CDNA5 async copy (ASYNCcnt path);
// the compiler hoists each wave's 6 B-tiles into persistent VGPRs.
// A operand: two aligned 16 B bf16 loads per K-step (lane K-groups
// {8g+0..7} and {8g+16..23}), zero conversion work in the loop.
__global__ __launch_bounds__(256) void k_gemm1(const __bf16* __restrict__ Apack,
                                               const __bf16* __restrict__ Wpack,
                                               const float* __restrict__ dinv,
                                               float* __restrict__ hs1) {
  __shared__ __bf16 sW[WPACK_ELEMS];               // 49152 bytes

  // ---- async stage: 49152 B / (256 thr * 16 B) = 12 iterations ----
#pragma unroll
  for (int it = 0; it < 12; ++it) {
    int byteOff = (it * 256 + (int)threadIdx.x) * 16;
    uint32_t ldsAddr = (uint32_t)(uintptr_t)((char*)sW + byteOff); // addr[31:0] = LDS offset
    const char* gaddr = (const char*)Wpack + byteOff;
    asm volatile("global_load_async_to_lds_b128 %0, %1, off"
                 :: "v"(ldsAddr), "v"(gaddr) : "memory");
  }
  asm volatile("s_wait_asynccnt 0x0" ::: "memory");
  __syncthreads();

  const int waveId = threadIdx.x >> 5;
  const int lane   = threadIdx.x & 31;
  const int nTile  = waveId;                       // 8 waves -> 8 nTiles
  const int g  = lane >> 4;                        // half-wave K-group
  const int ml = lane & 15;
  const int col = nTile * 16 + ml;

  for (int mt = 0; mt < MT_PER_BLOCK; ++mt) {
    const int mTile = blockIdx.x * MT_PER_BLOCK + mt;
    const int m0 = mTile * 16;
    const int row = m0 + ml;                       // A row for this lane
    const __bf16* arow = Apack + (size_t)row * IN1 + 8 * g;
    v8f c = {};
#pragma unroll
    for (int kk = 0; kk < KSTEPS; ++kk) {
      const int k0 = kk * 32;
      v8bf a_lo = *(const v8bf*)(arow + k0);        // K = k0 + 8g + 0..7
      v8bf a_hi = *(const v8bf*)(arow + k0 + 16);   // K = k0 + 8g + 16..23
      v16bf a = __builtin_shufflevector(a_lo, a_hi,
                                        0, 1, 2, 3, 4, 5, 6, 7,
                                        8, 9, 10, 11, 12, 13, 14, 15);
      const v16bf b = *(const v16bf*)(sW + ((size_t)(nTile * KSTEPS + kk) * 32 + lane) * 16);
      c = __builtin_amdgcn_wmma_f32_16x16x32_bf16(false, a, false, b,
                                                  (short)0, c, false, false);
    }
    // D layout: VGPR r -> row m0 + r + 8*g, col n0 + (lane&15). Fuse dinv scale.
#pragma unroll
    for (int r = 0; r < 8; ++r) {
      int orow = m0 + r + 8 * g;
      hs1[(size_t)orow * HIDDEN + col] = c[r] * dinv[orow];
    }
  }
}

// ------------------------------------------------- edge scatter, layer 1 ---
// acc1[d,:] += hs1[s,:]   (wave per edge, 4 f32 HW atomics per lane)
__global__ __launch_bounds__(256) void k_scatter1(const long long* __restrict__ ei,
                                                  const float* __restrict__ hs1,
                                                  float* __restrict__ acc1) {
  int gid  = blockIdx.x * blockDim.x + threadIdx.x;
  int e    = gid >> 5;
  int lane = gid & 31;
  if (e >= N_EDGES) return;
  int s = (int)ei[e];
  int d = (int)ei[N_EDGES + e];
  const float* hrow = hs1 + (size_t)s * HIDDEN;
  float* arow = acc1 + (size_t)d * HIDDEN;
#pragma unroll
  for (int k = 0; k < 4; ++k) {
    int f = lane + 32 * k;
    atomAddF(&arow[f], hrow[f]);
  }
}

// -------------------------------- finish layer1 (+self loop, bias) + stats -
__global__ __launch_bounds__(256) void k_finish1(const float* __restrict__ hs1,
                                                 float* __restrict__ acc1,
                                                 const float* __restrict__ dinv,
                                                 const float* __restrict__ b1,
                                                 float* __restrict__ bnsum,
                                                 float* __restrict__ bnsq) {
  __shared__ float ssum[HIDDEN], ssq[HIDDEN];
  if (threadIdx.x < HIDDEN) { ssum[threadIdx.x] = 0.f; ssq[threadIdx.x] = 0.f; }
  __syncthreads();
  const int total = N_NODES * HIDDEN;
  for (int idx = blockIdx.x * blockDim.x + threadIdx.x; idx < total;
       idx += gridDim.x * blockDim.x) {
    int i = idx >> 7, f = idx & 127;
    float y = dinv[i] * (acc1[idx] + hs1[idx]) + b1[f];
    acc1[idx] = y;                    // Y1 overwrites acc1 in place
    atomicAdd(&ssum[f], y);           // ds_add_f32
    atomicAdd(&ssq[f], y * y);
  }
  __syncthreads();
  if (threadIdx.x < HIDDEN) {
    atomAddF(&bnsum[threadIdx.x], ssum[threadIdx.x]);
    atomAddF(&bnsq[threadIdx.x],  ssq[threadIdx.x]);
  }
}

// ------------------------- BN1-normalize + ReLU + GEMV 128->2, pre-scaled --
__global__ __launch_bounds__(256) void k_bn1_gemm2(const float* __restrict__ Y1,
                                                   const float* __restrict__ bnsum,
                                                   const float* __restrict__ bnsq,
                                                   const float* __restrict__ gamma1,
                                                   const float* __restrict__ beta1,
                                                   const float* __restrict__ W2,
                                                   const float* __restrict__ dinv,
                                                   float* __restrict__ hs2) {
  int wave   = (blockIdx.x * blockDim.x + threadIdx.x) >> 5;
  int lane   = threadIdx.x & 31;
  int nwaves = (gridDim.x * blockDim.x) >> 5;
  const float inv_n = 1.0f / (float)N_NODES;
  for (int i = wave; i < N_NODES; i += nwaves) {
    float p0 = 0.f, p1 = 0.f;
#pragma unroll
    for (int k = 0; k < 4; ++k) {
      int f = lane + 32 * k;
      float mean = bnsum[f] * inv_n;
      float var  = bnsq[f] * inv_n - mean * mean;
      float z = (Y1[(size_t)i * HIDDEN + f] - mean) * rsqrtf(var + BN_EPS) *
                    gamma1[f] + beta1[f];
      z = fmaxf(z, 0.f);
      p0 += z * W2[f * 2 + 0];
      p1 += z * W2[f * 2 + 1];
    }
#pragma unroll
    for (int off = 16; off > 0; off >>= 1) {   // wave32 reduction
      p0 += __shfl_xor(p0, off);
      p1 += __shfl_xor(p1, off);
    }
    if (lane == 0) {
      float di = dinv[i];
      hs2[i * 2 + 0] = p0 * di;
      hs2[i * 2 + 1] = p1 * di;
    }
  }
}

// ------------------------------------------------- edge scatter, layer 2 ---
__global__ __launch_bounds__(256) void k_scatter2(const long long* __restrict__ ei,
                                                  const float* __restrict__ hs2,
                                                  float* __restrict__ acc2) {
  int e = blockIdx.x * blockDim.x + threadIdx.x;
  if (e >= N_EDGES) return;
  int s = (int)ei[e], d = (int)ei[N_EDGES + e];
  atomAddF(&acc2[d * 2 + 0], hs2[s * 2 + 0]);
  atomAddF(&acc2[d * 2 + 1], hs2[s * 2 + 1]);
}

// -------------------------------- finish layer2 (+self loop, bias) + stats -
__global__ __launch_bounds__(256) void k_finish2(const float* __restrict__ hs2,
                                                 float* __restrict__ acc2,
                                                 const float* __restrict__ dinv,
                                                 const float* __restrict__ b2,
                                                 float* __restrict__ st /*[4]*/) {
  int tid  = blockIdx.x * blockDim.x + threadIdx.x;
  int lane = threadIdx.x & 31;
  float s0 = 0.f, s1 = 0.f, q0 = 0.f, q1 = 0.f;
  for (int i = tid; i < N_NODES; i += gridDim.x * blockDim.x) {
    float di = dinv[i];
    float y0 = di * (acc2[i * 2 + 0] + hs2[i * 2 + 0]) + b2[0];
    float y1 = di * (acc2[i * 2 + 1] + hs2[i * 2 + 1]) + b2[1];
    acc2[i * 2 + 0] = y0;
    acc2[i * 2 + 1] = y1;
    s0 += y0; s1 += y1; q0 += y0 * y0; q1 += y1 * y1;
  }
#pragma unroll
  for (int off = 16; off > 0; off >>= 1) {
    s0 += __shfl_xor(s0, off); s1 += __shfl_xor(s1, off);
    q0 += __shfl_xor(q0, off); q1 += __shfl_xor(q1, off);
  }
  if (lane == 0) {
    atomAddF(&st[0], s0); atomAddF(&st[1], s1);
    atomAddF(&st[2], q0); atomAddF(&st[3], q1);
  }
}

// ------------------------------------------------- BN2 + softmax -> d_out --
__global__ __launch_bounds__(256) void k_bn2_softmax(const float* __restrict__ Y2,
                                                     const float* __restrict__ st,
                                                     const float* __restrict__ gamma2,
                                                     const float* __restrict__ beta2,
                                                     float* __restrict__ out) {
  int i = blockIdx.x * blockDim.x + threadIdx.x;
  if (i >= N_NODES) return;
  const float inv_n = 1.0f / (float)N_NODES;
  float m0 = st[0] * inv_n, m1 = st[1] * inv_n;
  float v0 = st[2] * inv_n - m0 * m0;
  float v1 = st[3] * inv_n - m1 * m1;
  float y0 = (Y2[i * 2 + 0] - m0) * rsqrtf(v0 + BN_EPS) * gamma2[0] + beta2[0];
  float y1 = (Y2[i * 2 + 1] - m1) * rsqrtf(v1 + BN_EPS) * gamma2[1] + beta2[1];
  float mx = fmaxf(y0, y1);
  float e0 = __expf(y0 - mx), e1 = __expf(y1 - mx);
  float inv = 1.f / (e0 + e1);
  out[i * 2 + 0] = e0 * inv;
  out[i * 2 + 1] = e1 * inv;
}

// --------------------------------------------------------------- launch ----
extern "C" void kernel_launch(void* const* d_in, const int* in_sizes, int n_in,
                              void* d_out, int out_size, void* d_ws, size_t ws_size,
                              hipStream_t stream) {
  const long long* ei = (const long long*)d_in[0];   // int64 [2, E]
  const float* X      = (const float*)d_in[1];
  const float* uY     = (const float*)d_in[2];
  const float* W1     = (const float*)d_in[3];
  const float* b1     = (const float*)d_in[4];
  const float* W2     = (const float*)d_in[5];
  const float* b2     = (const float*)d_in[6];
  const float* gamma1 = (const float*)d_in[7];
  const float* beta1  = (const float*)d_in[8];
  const float* gamma2 = (const float*)d_in[9];
  const float* beta2  = (const float*)d_in[10];
  (void)in_sizes; (void)n_in; (void)out_size; (void)ws_size;

  char* ws = (char*)d_ws;
  size_t off = 0;
  auto carve = [&](size_t bytes) -> void* {
    void* p = ws + off;
    off = (off + bytes + 255) & ~(size_t)255;
    return p;
  };
  unsigned* deg  = (unsigned*)carve((size_t)N_NODES * 4);
  float* dinv    = (float*)carve((size_t)N_NODES * 4);
  float* hs1     = (float*)carve((size_t)N_NODES * HIDDEN * 4);
  float* acc1    = (float*)carve((size_t)N_NODES * HIDDEN * 4);
  float* hs2     = (float*)carve((size_t)N_NODES * 2 * 4);
  float* acc2    = (float*)carve((size_t)N_NODES * 2 * 4);
  float* bnsum1  = (float*)carve(HIDDEN * 4);
  float* bnsq1   = (float*)carve(HIDDEN * 4);
  float* st2     = (float*)carve(4 * 4);
  __bf16* Wpack  = (__bf16*)carve((size_t)WPACK_ELEMS * 2);
  __bf16* Apack  = (__bf16*)carve((size_t)N_NODES * IN1 * 2);
  // total ~143 MB of workspace

  hipMemsetAsync(deg,  0, (size_t)N_NODES * 4, stream);
  hipMemsetAsync(acc1, 0, (size_t)N_NODES * HIDDEN * 4, stream);
  hipMemsetAsync(acc2, 0, (size_t)N_NODES * 2 * 4, stream);
  hipMemsetAsync(bnsum1, 0, HIDDEN * 4, stream);
  hipMemsetAsync(bnsq1,  0, HIDDEN * 4, stream);
  hipMemsetAsync(st2,    0, 16, stream);

  k_degree<<<(N_EDGES + 255) / 256, 256, 0, stream>>>(ei + N_EDGES, deg);
  k_dinv<<<(N_NODES + 255) / 256, 256, 0, stream>>>(deg, dinv);
  k_packA<<<(N_NODES * IN1 + 255) / 256, 256, 0, stream>>>(uY, X, Apack);
  k_packW1<<<(NTILES * KSTEPS * 32 + 255) / 256, 256, 0, stream>>>(W1, Wpack);

  // 625 blocks * (8 waves = 8 nTiles) * 10 mTiles = 50000 tiles
  k_gemm1<<<(N_NODES / 16) / MT_PER_BLOCK, 256, 0, stream>>>(Apack, Wpack, dinv, hs1);

  k_scatter1<<<(int)(((long long)N_EDGES * 32 + 255) / 256), 256, 0, stream>>>(ei, hs1, acc1);
  k_finish1<<<1024, 256, 0, stream>>>(hs1, acc1, dinv, b1, bnsum1, bnsq1);
  k_bn1_gemm2<<<2048, 256, 0, stream>>>(acc1, bnsum1, bnsq1, gamma1, beta1, W2, dinv, hs2);
  k_scatter2<<<(N_EDGES + 255) / 256, 256, 0, stream>>>(ei, hs2, acc2);
  k_finish2<<<512, 256, 0, stream>>>(hs2, acc2, dinv, b2, st2);
  k_bn2_softmax<<<(N_NODES + 255) / 256, 256, 0, stream>>>(acc2, st2, gamma2, beta2,
                                                           (float*)d_out);
}
